// SLU_GNN_62405874811574
// MI455X (gfx1250) — matrix operand order repaired and checked by reading, compiled, not run on hardware
//
#include <hip/hip_runtime.h>
#include <hip/hip_bf16.h>
#include <math.h>

// ---------------------------------------------------------------------------
// GatedGraphConv-style layer for MI455X (gfx1250, wave32, WMMA bf16->f32)
//   xw  = x @ W^T + b                  (bf16 WMMA; W staged in LDS via
//                                       global_load_async_to_lds_b128)
//   h   = segment_mean(xw[src] -> dst) (+ self loop)
//   h1  = GRU(h0=0, h)    fused GEMM + gates in regs; Wih staged in LDS
//   h2  = GRU(h1, xw)     fused dual-GEMM + gates in regs  -> d_out
// ---------------------------------------------------------------------------

typedef __attribute__((ext_vector_type(16))) __bf16 v16bf;
typedef __attribute__((ext_vector_type(8)))  float  v8f;

#define N_NODES   200000
#define N_EDGES   800000
#define IN_CH     256
#define OUT_CH    128
#define ROW_TILES (N_NODES / 16)   // 12500 (exact)

// ---- CDNA5 async global->LDS copy (16 bytes / lane, ASYNCcnt-tracked) ------
__device__ __forceinline__ void async_copy_b128(unsigned lds_byte_addr,
                                                const void* gaddr) {
  asm volatile("global_load_async_to_lds_b128 %0, %1, off"
               :: "v"(lds_byte_addr), "v"(gaddr) : "memory");
}
__device__ __forceinline__ void wait_async0() {
  asm volatile("s_wait_asynccnt 0x0" ::: "memory");
}

// ---- WMMA fragment loaders (contiguous k order, same in A and B) ------------------
// A matrix (16x32 bf16, MxK): lane l: m = l&15, group g = l>>4
//   lanes 0-15 hold K = ks..ks+15 contiguous; lanes 16-31 hold K = ks+16..ks+31
__device__ __forceinline__ v16bf load_frag_a(const __bf16* __restrict__ p,
                                             int row, int ld, int ks, int g) {
  union { v16bf v; uint4 q[2]; } u;
  const __bf16* rp = p + (size_t)row * ld + ks + g * 16;
  u.q[0] = *(const uint4*)(rp);
  u.q[1] = *(const uint4*)(rp + 8);
  return u.v;
}
// B matrix (32x16 bf16, KxN): lane l: n = l&15, group g = l>>4
//   lanes 0-15 hold K = ks..ks+15 contiguous; lanes 16-31 hold K = ks+16..ks+31
__device__ __forceinline__ v16bf load_frag_b(const __bf16* __restrict__ p,
                                             int row, int ld, int ks, int g) {
  union { v16bf v; uint4 q[2]; } u;
  const __bf16* rp = p + (size_t)row * ld + ks + g * 16;
  u.q[0] = *(const uint4*)(rp);
  u.q[1] = *(const uint4*)(rp + 8);
  return u.v;
}

__device__ __forceinline__ v8f wmma_bf16(v16bf a, v16bf b, v8f c) {
  // (neg_a, A, neg_b, B, c_mod, C, reuse_a, reuse_b)
  return __builtin_amdgcn_wmma_f32_16x16x32_bf16(false, a, false, b,
                                                 (short)0, c, false, false);
}

__device__ __forceinline__ float sigm(float x) { return 1.0f / (1.0f + __expf(-x)); }

// ---- utility kernels (all 4-wide vectorized; sizes are multiples of 4) -----
__global__ void k_zero_f32(float* __restrict__ p, int n4) {
  int i = blockIdx.x * blockDim.x + threadIdx.x;
  if (i < n4) ((float4*)p)[i] = (float4){0.f, 0.f, 0.f, 0.f};
}

__global__ void k_f32_to_bf16(const float* __restrict__ in,
                              __bf16* __restrict__ out, int n4) {
  int i = blockIdx.x * blockDim.x + threadIdx.x;
  if (i >= n4) return;
  float4 v = ((const float4*)in)[i];
  union { __bf16 h[4]; uint2 u; } o;
  o.h[0] = (__bf16)v.x; o.h[1] = (__bf16)v.y;
  o.h[2] = (__bf16)v.z; o.h[3] = (__bf16)v.w;
  ((uint2*)out)[i] = o.u;
}

// ---- xw = x @ W^T + b  (W staged in LDS; emit f32 + bf16 copies) -----------
__global__ void k_gemm_xw(const __bf16* __restrict__ X,     // [N_NODES, 256]
                          const __bf16* __restrict__ W,     // [128, 256]
                          const float*  __restrict__ bias,  // [128]
                          float*  __restrict__ outf,        // [N_NODES, 128]
                          __bf16* __restrict__ outb) {
  __shared__ __bf16 smW[OUT_CH * IN_CH];                    // 64 KB

  // async-stage W into LDS (4096 x 16B; 16 per thread with 256 threads)
  {
    unsigned lds0 = (unsigned)(size_t)&smW[0];
    const char* gw = (const char*)W;
#pragma unroll
    for (int it = 0; it < (OUT_CH * IN_CH * 2 / 16) / 256; ++it) {
      unsigned idx = threadIdx.x + it * 256;
      async_copy_b128(lds0 + idx * 16, gw + (size_t)idx * 16);
    }
    wait_async0();
  }
  __syncthreads();

  int wid = blockIdx.x * (blockDim.x >> 5) + (threadIdx.x >> 5);
  if (wid >= ROW_TILES) return;     // after barrier: no divergent sync
  int lane = threadIdx.x & 31;
  int lr = lane & 15, g = lane >> 4;
  int m0 = wid * 16;

  v8f acc[8];
#pragma unroll
  for (int t = 0; t < 8; ++t) acc[t] = (v8f){0.f,0.f,0.f,0.f,0.f,0.f,0.f,0.f};

  for (int ks = 0; ks < IN_CH; ks += 32) {
    v16bf a = load_frag_a(X, m0 + lr, IN_CH, ks, g);
#pragma unroll
    for (int t = 0; t < 8; ++t) {
      union { v16bf v; uint4 q[2]; } u;   // B frag from LDS (ds_load_b128)
      const __bf16* rp = &smW[(t * 16 + lr) * IN_CH + ks + g * 16];
      u.q[0] = *(const uint4*)(rp);
      u.q[1] = *(const uint4*)(rp + 8);
      acc[t] = wmma_bf16(a, u.v, acc[t]);
    }
  }
#pragma unroll
  for (int t = 0; t < 8; ++t) {
    int col = t * 16 + lr;
    float bv = bias[col];
#pragma unroll
    for (int i = 0; i < 8; ++i) {
      int m = m0 + i + 8 * g;               // D layout: VGPR i -> M = i + 8*g
      float v = acc[t][i] + bv;
      size_t idx = (size_t)m * OUT_CH + col;
      outf[idx] = v;
      outb[idx] = (__bf16)v;
    }
  }
}

// ---- edge scatter: agg[dst] += xw[src]; cnt[dst] += 1 ----------------------
__global__ void k_scatter(const long long* __restrict__ ei,   // [2, N_EDGES]
                          const float* __restrict__ xw,
                          float* __restrict__ agg,
                          float* __restrict__ cnt) {
  int e = blockIdx.x * (blockDim.x >> 5) + (threadIdx.x >> 5);
  if (e >= N_EDGES) return;
  int lane = threadIdx.x & 31;
  long long s = ei[e];
  long long d = ei[N_EDGES + e];
  float4 v = *(const float4*)(xw + (size_t)s * OUT_CH + lane * 4);
  float* ap = agg + (size_t)d * OUT_CH + lane * 4;
  atomicAdd(ap + 0, v.x);
  atomicAdd(ap + 1, v.y);
  atomicAdd(ap + 2, v.z);
  atomicAdd(ap + 3, v.w);
  if (lane == 0) atomicAdd(cnt + d, 1.0f);
}

// ---- h = (agg + xw_selfloop) / (cnt + 1)  -> bf16  (4-wide) ----------------
__global__ void k_finalize_h(const float* __restrict__ agg,
                             const float* __restrict__ xw,
                             const float* __restrict__ cnt,
                             __bf16* __restrict__ hb) {
  int i = blockIdx.x * blockDim.x + threadIdx.x;   // index of float4 group
  if (i >= (N_NODES * OUT_CH) / 4) return;
  int m = i >> 5;                        // (i*4)/128 : same node for all 4
  float rc = 1.0f / (cnt[m] + 1.0f);     // + self loop; always >= 1
  float4 a = ((const float4*)agg)[i];
  float4 xv = ((const float4*)xw)[i];
  union { __bf16 h[4]; uint2 u; } o;
  o.h[0] = (__bf16)((a.x + xv.x) * rc);
  o.h[1] = (__bf16)((a.y + xv.y) * rc);
  o.h[2] = (__bf16)((a.z + xv.z) * rc);
  o.h[3] = (__bf16)((a.w + xv.w) * rc);
  ((uint2*)hb)[i] = o.u;
}

// ---- GRU step 1: h1 = GRU(h0=0, h); gh reduces to b_hh ---------------------
__global__ void k_gru1(const __bf16* __restrict__ Hb,    // [N_NODES, 128]
                       const __bf16* __restrict__ Wih,   // [384, 128]
                       const float*  __restrict__ bih,   // [384]
                       const float*  __restrict__ bhh,   // [384]
                       float*  __restrict__ h1f,
                       __bf16* __restrict__ h1b) {
  __shared__ __bf16 smW[3 * OUT_CH * OUT_CH];             // 96 KB

  // async-stage Wih into LDS (6144 x 16B; 24 per thread with 256 threads)
  {
    unsigned lds0 = (unsigned)(size_t)&smW[0];
    const char* gw = (const char*)Wih;
#pragma unroll
    for (int it = 0; it < (3 * OUT_CH * OUT_CH * 2 / 16) / 256; ++it) {
      unsigned idx = threadIdx.x + it * 256;
      async_copy_b128(lds0 + idx * 16, gw + (size_t)idx * 16);
    }
    wait_async0();
  }
  __syncthreads();

  int wid = blockIdx.x * (blockDim.x >> 5) + (threadIdx.x >> 5);
  if (wid >= ROW_TILES) return;
  int lane = threadIdx.x & 31;
  int lr = lane & 15, g = lane >> 4;
  int m0 = wid * 16;

  v16bf a[4];
#pragma unroll
  for (int k = 0; k < 4; ++k) a[k] = load_frag_a(Hb, m0 + lr, OUT_CH, k * 32, g);

  const v8f z8 = (v8f){0.f,0.f,0.f,0.f,0.f,0.f,0.f,0.f};
  for (int t = 0; t < 8; ++t) {
    v8f ar = z8, az = z8, an = z8;
#pragma unroll
    for (int k = 0; k < 4; ++k) {
      int ks = k * 32;
#pragma unroll
      for (int gate = 0; gate < 3; ++gate) {
        union { v16bf v; uint4 q[2]; } u;  // B frag from LDS
        const __bf16* rp = &smW[(gate * 128 + t * 16 + lr) * OUT_CH + ks + g * 16];
        u.q[0] = *(const uint4*)(rp);
        u.q[1] = *(const uint4*)(rp + 8);
        if (gate == 0)      ar = wmma_bf16(a[k], u.v, ar);
        else if (gate == 1) az = wmma_bf16(a[k], u.v, az);
        else                an = wmma_bf16(a[k], u.v, an);
      }
    }
    int c = t * 16 + lr;
    float br  = bih[c]       + bhh[c];
    float bz  = bih[c + 128] + bhh[c + 128];
    float bni = bih[c + 256];
    float bnh = bhh[c + 256];
#pragma unroll
    for (int i = 0; i < 8; ++i) {
      int m = m0 + i + 8 * g;
      float r  = sigm(ar[i] + br);
      float z  = sigm(az[i] + bz);
      float nn = tanhf(an[i] + bni + r * bnh);
      float h1 = (1.0f - z) * nn;               // h_prev == 0
      size_t idx = (size_t)m * OUT_CH + c;
      h1f[idx] = h1;
      h1b[idx] = (__bf16)h1;
    }
  }
}

// ---- GRU step 2: h2 = GRU(h1, xw) -> d_out (weights via L0/L2) -------------
__global__ void k_gru2(const __bf16* __restrict__ Xwb,   // [N_NODES, 128]
                       const __bf16* __restrict__ H1b,   // [N_NODES, 128]
                       const float*  __restrict__ h1f,   // [N_NODES, 128]
                       const __bf16* __restrict__ Wih,   // [384, 128]
                       const __bf16* __restrict__ Whh,   // [384, 128]
                       const float*  __restrict__ bih,
                       const float*  __restrict__ bhh,
                       float* __restrict__ out) {
  int wid = blockIdx.x * (blockDim.x >> 5) + (threadIdx.x >> 5);
  if (wid >= ROW_TILES) return;
  int lane = threadIdx.x & 31;
  int lr = lane & 15, g = lane >> 4;
  int m0 = wid * 16;

  v16bf ax[4], ah[4];
#pragma unroll
  for (int k = 0; k < 4; ++k) {
    ax[k] = load_frag_a(Xwb, m0 + lr, OUT_CH, k * 32, g);
    ah[k] = load_frag_a(H1b, m0 + lr, OUT_CH, k * 32, g);
  }

  const v8f z8 = (v8f){0.f,0.f,0.f,0.f,0.f,0.f,0.f,0.f};
  for (int t = 0; t < 8; ++t) {
    v8f gr = z8, gz = z8, gn = z8;    // gi = xw @ Wih^T
    v8f hr = z8, hz = z8, hn = z8;    // gh = h1 @ Whh^T
#pragma unroll
    for (int k = 0; k < 4; ++k) {
      int ks = k * 32;
      v16bf bm[6];
      bm[0] = load_frag_b(Wih,       t * 16 + lr, OUT_CH, ks, g);
      bm[1] = load_frag_b(Wih, 128 + t * 16 + lr, OUT_CH, ks, g);
      bm[2] = load_frag_b(Wih, 256 + t * 16 + lr, OUT_CH, ks, g);
      bm[3] = load_frag_b(Whh,       t * 16 + lr, OUT_CH, ks, g);
      bm[4] = load_frag_b(Whh, 128 + t * 16 + lr, OUT_CH, ks, g);
      bm[5] = load_frag_b(Whh, 256 + t * 16 + lr, OUT_CH, ks, g);
      gr = wmma_bf16(ax[k], bm[0], gr);
      gz = wmma_bf16(ax[k], bm[1], gz);
      gn = wmma_bf16(ax[k], bm[2], gn);
      hr = wmma_bf16(ah[k], bm[3], hr);
      hz = wmma_bf16(ah[k], bm[4], hz);
      hn = wmma_bf16(ah[k], bm[5], hn);
    }
    int c = t * 16 + lr;
    float br  = bih[c]       + bhh[c];
    float bz  = bih[c + 128] + bhh[c + 128];
    float bni = bih[c + 256];
    float bnh = bhh[c + 256];
#pragma unroll
    for (int i = 0; i < 8; ++i) {
      int m = m0 + i + 8 * g;
      size_t idx = (size_t)m * OUT_CH + c;
      float r   = sigm(gr[i] + hr[i] + br);
      float z   = sigm(gz[i] + hz[i] + bz);
      float nn  = tanhf(gn[i] + bni + r * (hn[i] + bnh));
      float h1v = h1f[idx];
      out[idx] = (1.0f - z) * nn + z * h1v;
    }
  }
}

// ---------------------------------------------------------------------------
extern "C" void kernel_launch(void* const* d_in, const int* in_sizes, int n_in,
                              void* d_out, int out_size, void* d_ws, size_t ws_size,
                              hipStream_t stream) {
  const float*     x   = (const float*)d_in[0];
  const long long* ei  = (const long long*)d_in[1];   // int64 [2, N_EDGES]
  const float*     W   = (const float*)d_in[2];
  const float*     b   = (const float*)d_in[3];
  const float*     wih = (const float*)d_in[4];
  const float*     whh = (const float*)d_in[5];
  const float*     bih = (const float*)d_in[6];
  const float*     bhh = (const float*)d_in[7];
  float* out = (float*)d_out;

  // workspace layout (bytes)
  char* ws = (char*)d_ws;
  size_t o = 0;
  __bf16* xbf   = (__bf16*)(ws + o); o += (size_t)N_NODES * IN_CH * 2;   // 102.4 MB
  __bf16* Wbf   = (__bf16*)(ws + o); o += (size_t)OUT_CH * IN_CH * 2;    // 64 KB
  __bf16* wihbf = (__bf16*)(ws + o); o += (size_t)3 * OUT_CH * OUT_CH * 2;
  __bf16* whhbf = (__bf16*)(ws + o); o += (size_t)3 * OUT_CH * OUT_CH * 2;
  float*  xwf   = (float*) (ws + o); o += (size_t)N_NODES * OUT_CH * 4;  // 102.4 MB
  __bf16* xwbf  = (__bf16*)(ws + o); o += (size_t)N_NODES * OUT_CH * 2;  // 51.2 MB
  float*  agg   = (float*) (ws + o); o += (size_t)N_NODES * OUT_CH * 4;  // 102.4 MB
  float*  cnt   = (float*) (ws + o); o += (size_t)N_NODES * 4;
  __bf16* hbf   = (__bf16*)(ws + o); o += (size_t)N_NODES * OUT_CH * 2;
  float*  h1f   = (float*) (ws + o); o += (size_t)N_NODES * OUT_CH * 4;
  __bf16* h1bf  = (__bf16*)(ws + o); o += (size_t)N_NODES * OUT_CH * 2;

  const int TB = 256;
  // 1. fp32 -> bf16 conversions (4-wide)
  k_f32_to_bf16<<<(OUT_CH * IN_CH / 4 + TB - 1) / TB, TB, 0, stream>>>(W, Wbf, OUT_CH * IN_CH / 4);
  k_f32_to_bf16<<<(3 * OUT_CH * OUT_CH / 4 + TB - 1) / TB, TB, 0, stream>>>(wih, wihbf, 3 * OUT_CH * OUT_CH / 4);
  k_f32_to_bf16<<<(3 * OUT_CH * OUT_CH / 4 + TB - 1) / TB, TB, 0, stream>>>(whh, whhbf, 3 * OUT_CH * OUT_CH / 4);
  k_f32_to_bf16<<<(N_NODES * IN_CH / 4 + TB - 1) / TB, TB, 0, stream>>>(x, xbf, N_NODES * IN_CH / 4);

  // 2. zero accumulators (graph replays: must re-zero each call)
  k_zero_f32<<<(N_NODES * OUT_CH / 4 + TB - 1) / TB, TB, 0, stream>>>(agg, N_NODES * OUT_CH / 4);
  k_zero_f32<<<(N_NODES / 4 + TB - 1) / TB, TB, 0, stream>>>(cnt, N_NODES / 4);

  // 3. xw = x @ W^T + b   (1 wave per 16-row tile, 8 waves / block)
  int gemm_blocks = (ROW_TILES + 7) / 8;
  k_gemm_xw<<<gemm_blocks, TB, 0, stream>>>(xbf, Wbf, b, xwf, xwbf);

  // 4. edge scatter-add (1 wave per edge)
  k_scatter<<<(N_EDGES + 7) / 8, TB, 0, stream>>>(ei, xwf, agg, cnt);

  // 5. mean + self loop -> h (bf16)
  k_finalize_h<<<(N_NODES * OUT_CH / 4 + TB - 1) / TB, TB, 0, stream>>>(agg, xwf, cnt, hbf);

  // 6. GRU step 1 (fused GEMM + gates, Wih in LDS)
  k_gru1<<<gemm_blocks, TB, 0, stream>>>(hbf, wihbf, bih, bhh, h1f, h1bf);

  // 7. GRU step 2 (fused dual GEMM + gates) -> d_out
  k_gru2<<<gemm_blocks, TB, 0, stream>>>(xwbf, h1bf, h1f, wihbf, whhbf, bih, bhh, out);
}
